// Wav2VecPredictionsModel_2078764172096
// MI455X (gfx1250) — matrix operand order repaired and checked by reading, not compiled, not run on hardware
//
#include <hip/hip_runtime.h>

typedef __attribute__((ext_vector_type(16))) _Float16 v16h;
typedef __attribute__((ext_vector_type(8)))  float    v8f;

#define B_    16
#define D_    512
#define T_    1024
#define S_    12
#define NNEG  10
#define CCLS  11

// ---------------------------------------------------------------------------
// Prep 1: convert x (f32) -> f16, same [B, D, T] layout.
__global__ void k_cvt_x(const float* __restrict__ x, _Float16* __restrict__ xh, int n) {
    int i = blockIdx.x * 256 + threadIdx.x;
    if (i < n) xh[i] = (_Float16)x[i];
}

// ---------------------------------------------------------------------------
// Prep 2: W [k, o, s] (f32) -> Wt [s, k, o] (f16) so GEMM B-loads are coalesced.
__global__ void k_prep_w(const float* __restrict__ W, _Float16* __restrict__ Wt) {
    int idx = blockIdx.x * 256 + threadIdx.x;     // == output index
    int o = idx & 511;
    int k = (idx >> 9) & 511;
    int s = idx >> 18;
    Wt[idx] = (_Float16)W[((size_t)k * D_ + o) * S_ + s];
}

// ---------------------------------------------------------------------------
// Prep 3: y [b, f, t] -> yT [b, t, f] (tiled transpose; target columns become
// contiguous 2KB reads in stage 2; y fits in the 192MB L2 so gathers are cheap).
__global__ void k_transpose_y(const float* __restrict__ y, float* __restrict__ yT) {
    __shared__ float tile[32][33];
    int b  = blockIdx.z;
    int t0 = blockIdx.x * 32;
    int f0 = blockIdx.y * 32;
    int tx = threadIdx.x, ty = threadIdx.y;       // block (32, 8)
    #pragma unroll
    for (int r = 0; r < 4; ++r) {
        int f = f0 + ty + r * 8;
        tile[ty + r * 8][tx] = y[((size_t)b * D_ + f) * T_ + t0 + tx];
    }
    __syncthreads();
    #pragma unroll
    for (int r = 0; r < 4; ++r) {
        int t = t0 + ty + r * 8;
        yT[((size_t)b * T_ + t) * D_ + f0 + tx] = tile[tx][ty + r * 8];
    }
}

// ---------------------------------------------------------------------------
// A-matrix K swizzle: fragment element j needs K = [0..7,16..23] (lanes 0-15)
// and [8..15,24..31] (lanes 16-31). Storing K octets in order 0,2,1,3 makes
// each lane's v16h fragment a contiguous 32-byte run in LDS.
__device__ __forceinline__ int a_swz(int k) {
    int oct = k >> 3;                       // 0..3
    return (oct == 1 || oct == 2) ? (k ^ 24) : k;
}

// ---------------------------------------------------------------------------
// Stage 1: xp[m, s, o] = sum_k xh[b,k,t] * Wt[s,k,o] + bias[o],  m = b*T + t.
// 128x128 tile per block; 8 waves, each owns a 32x64 region = 2x4 WMMA tiles.
// f16 inputs, f32 accumulation via v_wmma_f32_16x16x32_f16 (wave32).
// Fragments are contiguous 32B runs in LDS -> ds_load_b128 pairs, no repack.
__global__ __launch_bounds__(256) void k_gemm(const _Float16* __restrict__ xh,
                                              const _Float16* __restrict__ Wt,
                                              const float* __restrict__ bias,
                                              float* __restrict__ xp) {
    // Row stride 40 halves = 80B: 16B-aligned rows for b128 LDS reads,
    // bank stride 20 dwords (conflict-light on 64 banks).
    __shared__ __align__(16) _Float16 As[128][40];   // [m-row][k]
    __shared__ __align__(16) _Float16 Bs[128][40];   // [o-col][k]

    const int tid  = threadIdx.x;
    const int m0   = blockIdx.x * 128;
    const int b    = m0 >> 10;         // 128 | 1024 so one b per tile
    const int t0   = m0 & 1023;
    const int s    = blockIdx.y >> 2;
    const int o0   = (blockIdx.y & 3) * 128;

    const int w      = tid >> 5;
    const int lane   = tid & 31;
    const int half   = lane >> 4;      // lane group 0-15 vs 16-31
    const int ln     = lane & 15;
    const int wave_m = (w & 3) * 32;
    const int wave_n = (w >> 2) * 64;

    // Init accumulators with bias (C/D layout: per-lane N is fixed = ln).
    v8f acc[2][4];
    #pragma unroll
    for (int tn = 0; tn < 4; ++tn) {
        float bv = bias[o0 + wave_n + tn * 16 + ln];
        #pragma unroll
        for (int tm = 0; tm < 2; ++tm)
            #pragma unroll
            for (int r = 0; r < 8; ++r) acc[tm][tn][r] = bv;
    }

    const size_t xbase = (size_t)b * D_ * T_ + t0;
    const size_t wbase = (size_t)s * D_ * D_ + o0;

    for (int kk = 0; kk < D_; kk += 32) {
        __syncthreads();
        if (kk + 32 < D_) {  // prefetch next K-chunk (global_prefetch_b8, near)
            __builtin_prefetch(xh + xbase + (size_t)(kk + 32 + (tid >> 7)) * T_ + (tid & 127), 0, 3);
            __builtin_prefetch(Wt + wbase + (size_t)(kk + 32 + (tid >> 7)) * D_ + (tid & 127), 0, 3);
        }
        #pragma unroll
        for (int it = 0; it < 16; ++it) {
            int lin = it * 256 + tid;
            int rc  = lin & 127;       // coalesced over t / o
            int kc  = lin >> 7;        // 0..31
            As[rc][kc]        = xh[xbase + (size_t)(kk + kc) * T_ + rc];
            Bs[rc][kc]        = Wt[wbase + (size_t)(kk + kc) * D_ + rc];
        }
        __syncthreads();

        // Contiguous 32B fragment loads (two ds_load_b128 each).
        v16h a[2], bfr[4];
        #pragma unroll
        for (int tm = 0; tm < 2; ++tm)
            a[tm] = *(const v16h*)&As[wave_m + tm * 16 + ln][half * 16];
        #pragma unroll
        for (int tn = 0; tn < 4; ++tn)
            bfr[tn] = *(const v16h*)&Bs[wave_n + tn * 16 + ln][half * 16];

        #pragma unroll
        for (int tm = 0; tm < 2; ++tm)
            #pragma unroll
            for (int tn = 0; tn < 4; ++tn)
                acc[tm][tn] = __builtin_amdgcn_wmma_f32_16x16x32_f16(
                    false, a[tm], false, bfr[tn],
                    (short)0, acc[tm][tn], false, false);
    }

    // Store D: VGPR r holds M = r + 8*half, N = lane&15.
    #pragma unroll
    for (int tm = 0; tm < 2; ++tm) {
        #pragma unroll
        for (int tn = 0; tn < 4; ++tn) {
            int o = o0 + wave_n + tn * 16 + ln;
            #pragma unroll
            for (int r = 0; r < 8; ++r) {
                int m = m0 + wave_m + tm * 16 + r + 8 * half;
                xp[((size_t)m * S_ + s) * D_ + o] = acc[tm][tn][r];
            }
        }
    }
}

// ---------------------------------------------------------------------------
// Stage 2: one block per (b, tau). 11 waves = one wave per class c.
// p[c,b,t,i] = dot_f( xp[b,:,t,i], target_c[b,:,t+i+1] ), tau = t+i+1.
// Targets (pos + 10 gathered negs) staged once in LDS; each of the 12
// xp columns staged once and shared by all 11 waves. Labels written too.
__global__ __launch_bounds__(352) void k_score(const float* __restrict__ xp,
                                               const float* __restrict__ yT,
                                               const int* __restrict__ negi,
                                               float* __restrict__ out) {
    __shared__ float tgt[CCLS][D_];
    __shared__ float xcol[D_];

    const int tau  = blockIdx.x;
    const int b    = blockIdx.y;
    const int tid  = threadIdx.x;
    const int w    = tid >> 5;        // class id, 0..10
    const int lane = tid & 31;

    for (int idx = tid; idx < CCLS * D_; idx += 352) {
        int c = idx >> 9, f = idx & 511;
        int tcol = (c == 0) ? tau
                            : negi[((size_t)b * NNEG + (c - 1)) * T_ + tau];
        tgt[c][f] = yT[((size_t)b * T_ + tcol) * D_ + f];
    }
    __syncthreads();

    const long long P_TOTAL =
        (long long)CCLS * B_ * ((long long)S_ * T_ - (S_ * (S_ + 1)) / 2);

    for (int i = 0; i < S_; ++i) {
        int t = tau - 1 - i;          // block-uniform guard; barriers stay uniform
        __syncthreads();
        if (t >= 0) {
            for (int f = tid; f < D_; f += 352)
                xcol[f] = xp[(((size_t)b * T_ + t) * S_ + i) * D_ + f];
        }
        __syncthreads();
        if (t >= 0) {
            float p = 0.f;
            for (int f = lane; f < D_; f += 32)
                p += xcol[f] * tgt[w][f];
            #pragma unroll
            for (int off = 16; off > 0; off >>= 1)
                p += __shfl_xor(p, off, 32);
            if (lane == 0) {
                int offi = i + 1;
                long long Tm   = T_ - offi;
                long long base = (long long)CCLS * B_ *
                                 ((long long)i * T_ - (long long)i * (i + 1) / 2);
                long long idx  = base + (long long)w * B_ * Tm + (long long)b * Tm + t;
                out[idx]           = p;
                out[P_TOTAL + idx] = (w == 0) ? 1.0f : 0.0f;
            }
        }
    }
}

// ---------------------------------------------------------------------------
extern "C" void kernel_launch(void* const* d_in, const int* in_sizes, int n_in,
                              void* d_out, int out_size, void* d_ws, size_t ws_size,
                              hipStream_t stream) {
    const float* x    = (const float*)d_in[0];   // [16, 512, 1024]
    const float* y    = (const float*)d_in[1];   // [16, 512, 1024]
    const float* W    = (const float*)d_in[2];   // [512, 512, 12]
    const float* bias = (const float*)d_in[3];   // [512]
    const int*   negi = (const int*)d_in[4];     // [16, 10*1024]
    float* out = (float*)d_out;

    // Workspace layout (total ~438 MB):
    //   xp  f32  [16384, 12, 512]  = 402,653,184 B
    //   xh  f16  [16, 512, 1024]   =  16,777,216 B
    //   Wt  f16  [12, 512, 512]    =   6,291,456 B
    //   yT  f32  [16, 1024, 512]   =  33,554,432 B
    char* ws = (char*)d_ws;
    float*    xp = (float*)ws;
    _Float16* xh = (_Float16*)(ws + 402653184ULL);
    _Float16* Wt = (_Float16*)(ws + 419430400ULL);
    float*    yT = (float*)(ws + 425721856ULL);

    const int NX = B_ * D_ * T_;       // 8,388,608
    const int NW = D_ * D_ * S_;       // 3,145,728

    k_cvt_x<<<NX / 256, 256, 0, stream>>>(x, xh, NX);
    k_prep_w<<<NW / 256, 256, 0, stream>>>(W, Wt);
    k_transpose_y<<<dim3(T_ / 32, D_ / 32, B_), dim3(32, 8), 0, stream>>>(y, yT);
    k_gemm<<<dim3((B_ * T_) / 128, (D_ / 128) * S_), 256, 0, stream>>>(xh, Wt, bias, xp);
    k_score<<<dim3(T_, B_), 352, 0, stream>>>(xp, yT, negi, out);
}
